// GNNBackbone_1941325218075
// MI455X (gfx1250) — matrix operand order repaired and checked by reading, not compiled, not run on hardware
//
#include <hip/hip_runtime.h>
#include <hip/hip_bf16.h>

#define BATCH 256
#define DNODES 1000
#define FVD 300
#define EMBD 128
#define HID 128
#define NEDGE 32000

typedef __attribute__((ext_vector_type(16))) __bf16 v16bf;
typedef __attribute__((ext_vector_type(8)))  __bf16 v8bf;
typedef __attribute__((ext_vector_type(8)))  float  v8f;

__device__ __forceinline__ float relu_f(float x) { return x > 0.f ? x : 0.f; }

// Issue a per-lane 16B async copy global -> LDS (ASYNCcnt-tracked DMA).
__device__ __forceinline__ void async_copy_b128(uint32_t lds_off,
                                                const void* gptr) {
  uint64_t ga = (uint64_t)(uintptr_t)gptr;
  asm volatile("global_load_async_to_lds_b128 %0, %1, off"
               :
               : "v"(lds_off), "v"(ga)
               : "memory");
}
__device__ __forceinline__ void async_wait0() {
  asm volatile("s_wait_asynccnt 0x0" ::: "memory");
}
__device__ __forceinline__ uint32_t lds_addr_of(const void* p) {
  // generic shared pointer = aperture | lds_offset; low 32 bits are the offset
  return (uint32_t)(uintptr_t)p;
}

// ---- WMMA fragment loaders (layouts per CDNA5 ISA 7.12.2) --------------------
// A (16x32 bf16): lanes 0-15 -> row M=lane, K {kb..kb+7, kb+16..kb+23};
//                 lanes 16-31 -> row M=lane-16, K {kb+8..kb+15, kb+24..kb+31}.
__device__ __forceinline__ v16bf load_a_frag(const __bf16* src, int rowBase,
                                             int kbase, int lane) {
  int l15 = lane & 15;
  int half = lane >> 4;
  const __bf16* p = src + (rowBase + l15) * HID + kbase + half * 8;
  v8bf lo = *(const v8bf*)p;
  v8bf hi = *(const v8bf*)(p + 16);
  return __builtin_shufflevector(lo, hi, 0, 1, 2, 3, 4, 5, 6, 7,
                                 8, 9, 10, 11, 12, 13, 14, 15);
}

// B (32x16 bf16): lanes 0-15 -> col N=lane, K {kb..kb+7, kb+16..kb+23};
// lanes 16-31 -> col N=lane-16, K {kb+8..kb+15, kb+24..kb+31}. Wt stored as Wt[k][n].
__device__ __forceinline__ v16bf load_b_frag(const __bf16* wt, int kbase,
                                             int ntile, int lane) {
  int l15 = lane & 15;
  int half = lane >> 4;
  const __bf16* p = wt + (kbase + half * 8) * HID + ntile * 16 + l15;
  v16bf r;
#pragma unroll
  for (int s = 0; s < 8; ++s) {
    r[s] = p[s * HID];
    r[8 + s] = p[(16 + s) * HID];
  }
  return r;
}

// One wave computes rows [rowBase, rowBase+16) x 128 of  h @ W^T  with
// 4 K-chunks x 8 N-tiles = 32 v_wmma_f32_16x16x32_bf16.  hsrc and wt both LDS.
__device__ __forceinline__ void wmma_rows16(const __bf16* hsrc, int rowBase,
                                            const __bf16* wt, v8f acc[8]) {
  int lane = threadIdx.x & 31;
  v8f z = {0.f, 0.f, 0.f, 0.f, 0.f, 0.f, 0.f, 0.f};
#pragma unroll
  for (int nt = 0; nt < 8; ++nt) acc[nt] = z;
#pragma unroll
  for (int kc = 0; kc < 4; ++kc) {
    v16bf a = load_a_frag(hsrc, rowBase, kc * 32, lane);
#pragma unroll
    for (int nt = 0; nt < 8; ++nt) {
      v16bf bm = load_b_frag(wt, kc * 32, nt, lane);
      acc[nt] = __builtin_amdgcn_wmma_f32_16x16x32_bf16(
          false, a, false, bm, (short)0, acc[nt], false, false);
    }
  }
}

// ---- prep kernels ------------------------------------------------------------
__global__ void gnn_zero_out(float* out) {
  int i = blockIdx.x * blockDim.x + threadIdx.x;
  if (i < BATCH * HID) out[i] = 0.f;
}

// emb[d][f] = sum_k V[d][k] * Wemb[f][k] + bemb[f]
__global__ void gnn_prep_emb(const float* __restrict__ V,
                             const float* __restrict__ Wemb,
                             const float* __restrict__ bemb,
                             float* __restrict__ emb) {
  __shared__ float sV[FVD];
  int d = blockIdx.x, f = threadIdx.x;
  for (int k = f; k < FVD; k += 128) sV[k] = V[d * FVD + k];
  __syncthreads();
  float s = bemb[f];
  for (int k = 0; k < FVD; ++k) s += sV[k] * Wemb[f * FVD + k];
  emb[d * EMBD + f] = s;
}

// embW0[d][f] = sum_k emb[d][k] * W0[f][k]   (bias applied later)
__global__ void gnn_prep_embw0(const float* __restrict__ emb,
                               const float* __restrict__ W0,
                               float* __restrict__ embW0) {
  __shared__ float sE[EMBD];
  int d = blockIdx.x, f = threadIdx.x;
  sE[f] = emb[d * EMBD + f];
  __syncthreads();
  float s = 0.f;
  for (int k = 0; k < EMBD; ++k) s += sE[k] * W0[f * EMBD + k];
  embW0[d * HID + f] = s;
}

// Wt[k][n] = bf16(W[n][k]) for W1 (blocks 0..127) and W2 (blocks 128..255)
__global__ void gnn_prep_wt(const float* __restrict__ W1,
                            const float* __restrict__ W2,
                            __bf16* __restrict__ wt1, __bf16* __restrict__ wt2) {
  int blk = blockIdx.x;
  int which = blk >> 7;
  int k = blk & 127;
  int n = threadIdx.x;
  const float* W = which ? W2 : W1;
  __bf16* wt = which ? wt2 : wt1;
  wt[k * HID + n] = (__bf16)W[n * HID + k];
}

// ---- main fused MLP for batches 1..255 (no graph coupling; deg==1) -----------
__global__ __launch_bounds__(256) void gnn_fused_mlp(
    const float* __restrict__ Xnum, const float* __restrict__ Xcat,
    const float* __restrict__ embW0, const __bf16* __restrict__ wt1,
    const __bf16* __restrict__ wt2, const float* __restrict__ b0v,
    const float* __restrict__ b1v, const float* __restrict__ b2v,
    float* __restrict__ out) {
  __shared__ __align__(16) __bf16 hA[128 * HID];
  __shared__ __align__(16) __bf16 hB[128 * HID];
  __shared__ __align__(16) __bf16 wS[2 * 128 * HID];  // wt1 | wt2 (64 KB)
  int b = (blockIdx.x >> 3) + 1;  // batches 1..255
  int tile = blockIdx.x & 7;      // 8 tiles of 128 d-rows
  int tid = threadIdx.x;

  // Kick off async DMA of both weight matrices into LDS (overlaps stage below).
  {
    uint32_t lbase = lds_addr_of(&wS[0]);
    const char* g1 = (const char*)wt1;
    const char* g2 = (const char*)wt2;
#pragma unroll
    for (int i = 0; i < 8; ++i) {
      uint32_t off = (uint32_t)(tid + i * 256) * 16;  // 8*256*16B = 32 KB each
      async_copy_b128(lbase + off, g1 + off);
      async_copy_b128(lbase + 32768 + off, g2 + off);
    }
  }

  // Stage h1 = relu(x[b,d]*embW0[d] + b0) as bf16 in LDS (zero-pad d>=1000).
  for (int idx = tid; idx < 128 * HID; idx += 256) {
    int row = idx >> 7, col = idx & 127;
    int d = tile * 128 + row;
    float v = 0.f;
    if (d < DNODES) {
      float xv = (d < 800) ? Xnum[b * 800 + d] : Xcat[b * 200 + (d - 800)];
      v = relu_f(xv * embW0[d * HID + col] + b0v[col]);
    }
    hA[idx] = (__bf16)v;
  }
  async_wait0();
  __syncthreads();

  int wave = tid >> 5, lane = tid & 31;
  int l15 = lane & 15, half = lane >> 4;
  int rowBase = wave * 16;
  v8f acc[8];
  float bc[8];

  // Layer 2: h2 = relu(h1 @ W1^T + b1) -> LDS
  wmma_rows16(hA, rowBase, &wS[0], acc);
#pragma unroll
  for (int nt = 0; nt < 8; ++nt) bc[nt] = b1v[nt * 16 + l15];
#pragma unroll
  for (int nt = 0; nt < 8; ++nt)
#pragma unroll
    for (int i = 0; i < 8; ++i) {
      int row = rowBase + (half ? 8 + i : i);  // C layout: VGPR i -> M=i / M=8+i
      int col = nt * 16 + l15;
      hB[row * HID + col] = (__bf16)relu_f(acc[nt][i] + bc[nt]);
    }
  __syncthreads();

  // Layer 3: h3 = relu(h2 @ W2^T + b2), masked column-sum into out (mean).
  wmma_rows16(hB, rowBase, &wS[128 * HID], acc);
#pragma unroll
  for (int nt = 0; nt < 8; ++nt) bc[nt] = b2v[nt * 16 + l15];
#pragma unroll
  for (int nt = 0; nt < 8; ++nt) {
    float s = 0.f;
#pragma unroll
    for (int i = 0; i < 8; ++i) {
      int row = rowBase + (half ? 8 + i : i);
      int d = tile * 128 + row;
      float v = relu_f(acc[nt][i] + bc[nt]);
      s += (d < DNODES) ? v : 0.f;
    }
    s += __shfl_xor(s, 16, 32);  // combine M 0..7 with M 8..15 halves
    if (half == 0)
      atomicAdd(&out[b * HID + nt * 16 + l15], s * (1.0f / (float)DNODES));
  }
}

// ---- batch-0 graph path (1000 coupled nodes; tiny) ---------------------------
__global__ void gnn_b0_hw0(const float* __restrict__ Xnum,
                           const float* __restrict__ Xcat,
                           const float* __restrict__ embW0,
                           float* __restrict__ hw) {
  int d = blockIdx.x, f = threadIdx.x;
  float xv = (d < 800) ? Xnum[d] : Xcat[d - 800];
  hw[d * HID + f] = xv * embW0[d * HID + f];
}

__global__ void gnn_deg_init(float* deg) {
  int t = blockIdx.x * blockDim.x + threadIdx.x;
  if (t < DNODES) deg[t] = 1.f;
}

__global__ void gnn_deg_edges(const int* __restrict__ ei, float* deg) {
  int e = blockIdx.x * blockDim.x + threadIdx.x;
  if (e < NEDGE) atomicAdd(&deg[ei[NEDGE + e]], 1.f);
}

// agg[t] = hw[t] * (1/deg[t])  (self-loop term; overwrites agg)
__global__ void gnn_agg_init(const float* __restrict__ hw,
                             const float* __restrict__ deg,
                             float* __restrict__ agg) {
  int t = blockIdx.x, f = threadIdx.x;
  agg[t * HID + f] = hw[t * HID + f] * (1.f / deg[t]);
}

__global__ void gnn_agg_edges(const float* __restrict__ hw,
                              const float* __restrict__ deg,
                              const int* __restrict__ ei,
                              float* __restrict__ agg) {
  int e = blockIdx.x, f = threadIdx.x;
  int s = ei[e], t = ei[NEDGE + e];
  float coef = rsqrtf(deg[s]) * rsqrtf(deg[t]);
  atomicAdd(&agg[t * HID + f], hw[s * HID + f] * coef);
}

__global__ void gnn_b0_post(const float* __restrict__ agg,
                            const float* __restrict__ bias,
                            float* __restrict__ h) {
  int t = blockIdx.x, f = threadIdx.x;
  h[t * HID + f] = relu_f(agg[t * HID + f] + bias[f]);
}

// hw = h @ W^T over the 1000 batch-0 nodes (WMMA, bf16 staged in LDS)
__global__ __launch_bounds__(256) void gnn_b0_gemm(
    const float* __restrict__ hin, const __bf16* __restrict__ wt,
    float* __restrict__ hwout) {
  __shared__ __align__(16) __bf16 hA[128 * HID];
  __shared__ __align__(16) __bf16 wS[128 * HID];
  int tile = blockIdx.x;
  int tid = threadIdx.x;
  {
    uint32_t lbase = lds_addr_of(&wS[0]);
    const char* g = (const char*)wt;
#pragma unroll
    for (int i = 0; i < 8; ++i) {
      uint32_t off = (uint32_t)(tid + i * 256) * 16;
      async_copy_b128(lbase + off, g + off);
    }
  }
  for (int idx = tid; idx < 128 * HID; idx += 256) {
    int row = idx >> 7, col = idx & 127;
    int d = tile * 128 + row;
    hA[idx] = (__bf16)((d < DNODES) ? hin[d * HID + col] : 0.f);
  }
  async_wait0();
  __syncthreads();
  int wave = tid >> 5, lane = tid & 31;
  int l15 = lane & 15, half = lane >> 4;
  v8f acc[8];
  wmma_rows16(hA, wave * 16, wS, acc);
#pragma unroll
  for (int nt = 0; nt < 8; ++nt)
#pragma unroll
    for (int i = 0; i < 8; ++i) {
      int row = wave * 16 + (half ? 8 + i : i);
      int d = tile * 128 + row;
      if (d < DNODES) hwout[d * HID + nt * 16 + l15] = acc[nt][i];
    }
}

__global__ void gnn_b0_mean(const float* __restrict__ h, float* __restrict__ out) {
  int f = threadIdx.x;
  float s = 0.f;
  for (int d = 0; d < DNODES; ++d) s += h[d * HID + f];
  out[f] = s * (1.f / (float)DNODES);
}

// ---- workspace layout (bytes) ------------------------------------------------
static constexpr size_t OFF_EMB   = 0;        // 1000*128*4 = 512000
static constexpr size_t OFF_EMBW0 = 512000;   // 512000
static constexpr size_t OFF_HW    = 1024000;  // 512000
static constexpr size_t OFF_AGG   = 1536000;  // 512000
static constexpr size_t OFF_H     = 2048000;  // 512000
static constexpr size_t OFF_DEG   = 2560000;  // 4000
static constexpr size_t OFF_WT1   = 2564096;  // 128*128*2 = 32768
static constexpr size_t OFF_WT2   = 2596864;  // 32768 -> end ~2.51 MB

extern "C" void kernel_launch(void* const* d_in, const int* in_sizes, int n_in,
                              void* d_out, int out_size, void* d_ws, size_t ws_size,
                              hipStream_t stream) {
  (void)in_sizes; (void)n_in; (void)out_size; (void)ws_size;
  const float* Xnum = (const float*)d_in[0];
  const float* Xcat = (const float*)d_in[1];
  const float* V    = (const float*)d_in[2];
  const int*   ei   = (const int*)d_in[3];
  const float* Wemb = (const float*)d_in[4];
  const float* bemb = (const float*)d_in[5];
  const float* W0   = (const float*)d_in[6];
  const float* b0v  = (const float*)d_in[7];
  const float* W1   = (const float*)d_in[8];
  const float* b1v  = (const float*)d_in[9];
  const float* W2   = (const float*)d_in[10];
  const float* b2v  = (const float*)d_in[11];
  float* out = (float*)d_out;
  char* ws = (char*)d_ws;
  float*  emb   = (float*)(ws + OFF_EMB);
  float*  embW0 = (float*)(ws + OFF_EMBW0);
  float*  hw    = (float*)(ws + OFF_HW);
  float*  agg   = (float*)(ws + OFF_AGG);
  float*  hbuf  = (float*)(ws + OFF_H);
  float*  deg   = (float*)(ws + OFF_DEG);
  __bf16* wt1   = (__bf16*)(ws + OFF_WT1);
  __bf16* wt2   = (__bf16*)(ws + OFF_WT2);

  gnn_zero_out<<<(BATCH * HID + 255) / 256, 256, 0, stream>>>(out);
  gnn_prep_emb<<<DNODES, 128, 0, stream>>>(V, Wemb, bemb, emb);
  gnn_prep_embw0<<<DNODES, 128, 0, stream>>>(emb, W0, embW0);
  gnn_prep_wt<<<256, 128, 0, stream>>>(W1, W2, wt1, wt2);

  // Batches 1..255: fully fused 3-layer MLP + mean (WMMA, LDS-resident h,
  // weights async-DMA'd to LDS).
  gnn_fused_mlp<<<255 * 8, 256, 0, stream>>>(Xnum, Xcat, embW0, wt1, wt2,
                                             b0v, b1v, b2v, out);

  // Batch 0: GCN aggregation over the 1000 edge-referenced nodes.
  gnn_b0_hw0<<<DNODES, 128, 0, stream>>>(Xnum, Xcat, embW0, hw);
  gnn_deg_init<<<(DNODES + 127) / 128, 128, 0, stream>>>(deg);
  gnn_deg_edges<<<(NEDGE + 255) / 256, 256, 0, stream>>>(ei, deg);

  // layer 0 aggregation + bias/relu
  gnn_agg_init<<<DNODES, 128, 0, stream>>>(hw, deg, agg);
  gnn_agg_edges<<<NEDGE, 128, 0, stream>>>(hw, deg, ei, agg);
  gnn_b0_post<<<DNODES, 128, 0, stream>>>(agg, b0v, hbuf);
  // layer 1
  gnn_b0_gemm<<<8, 256, 0, stream>>>(hbuf, wt1, hw);
  gnn_agg_init<<<DNODES, 128, 0, stream>>>(hw, deg, agg);
  gnn_agg_edges<<<NEDGE, 128, 0, stream>>>(hw, deg, ei, agg);
  gnn_b0_post<<<DNODES, 128, 0, stream>>>(agg, b1v, hbuf);
  // layer 2
  gnn_b0_gemm<<<8, 256, 0, stream>>>(hbuf, wt2, hw);
  gnn_agg_init<<<DNODES, 128, 0, stream>>>(hw, deg, agg);
  gnn_agg_edges<<<NEDGE, 128, 0, stream>>>(hw, deg, ei, agg);
  gnn_b0_post<<<DNODES, 128, 0, stream>>>(agg, b2v, hbuf);

  gnn_b0_mean<<<1, 128, 0, stream>>>(hbuf, out);
}
